// SO2EquivariantGraphAttention_64372969832680
// MI455X (gfx1250) — matrix operand order repaired and checked by reading, not compiled, not run on hardware
//
#include <hip/hip_runtime.h>

// ---------------------------------------------------------------------------
// SO2 equivariant graph attention for MI455X (gfx1250, wave32, WMMA bf16)
// All large matmuls -> v_wmma_f32_16x16x32_bf16 (fp32 accumulate).
// Fragments loaded from LDS as b128 (A row-major padded, B transposed).
// GEMM specialized: M%128==0, N%64==0, K%32==0, lda/ldb%8==0 (all call sites).
// ---------------------------------------------------------------------------

typedef unsigned short u16;
typedef unsigned int   u32;
typedef unsigned long long u64;

typedef __attribute__((ext_vector_type(16))) __bf16 v16bf;
typedef __attribute__((ext_vector_type(8)))  float  v8f;

union BFrag { v16bf v; uint4 q[2]; u16 h[16]; };
union AccF  { v8f   v; float f[8]; };
union U4    { uint4 q; u16 h[8]; };

// problem constants
enum {
  NNODE = 1024, NE = 8192, NCH = 49, CCH = 128, HH = 64, VV = 128,
  MRED = 29, RADTOT = 4608, X0N = 1024, EXTRAC = 576, ALPHACH = 512
};

__constant__ int PM[29] = {0,2,6,11,16,21,26, 3,7,12,17,22,27,
                           1,5,10,15,20,25, 8,13,18,23,28, 4,9,14,19,24};
__constant__ int PL[29] = {0,13,1,7,24,14,2,8,19,25,15,3,9,20,26,
                           16,4,10,21,27,17,5,11,22,28,18,6,12,23};

__device__ __forceinline__ u16 f2bf(float f) {
  u32 u = __float_as_uint(f);
  u32 r = u + 0x7FFFu + ((u >> 16) & 1u);
  return (u16)(r >> 16);
}
__device__ __forceinline__ float bf2f(u16 h) { return __uint_as_float(((u32)h) << 16); }
__device__ __forceinline__ float siluf(float x) { return x / (1.0f + __expf(-x)); }

// A fragment: needs k = kb..kb+15 contiguous from a row-major row -> 2x b128
__device__ __forceinline__ void lda_frag(BFrag& f, const u16* row, int kb) {
  f.q[0] = *(const uint4*)(row + kb);
  f.q[1] = *(const uint4*)(row + kb + 8);
}
// B fragment: with transposed tile Bt[n][k], needs k = kh..kh+15 contiguous -> 2x b128
__device__ __forceinline__ void ldb_frag(BFrag& f, const u16* rowT, int kh) {
  f.q[0] = *(const uint4*)(rowT + kh);
  f.q[1] = *(const uint4*)(rowT + kh + 8);
}

// ---------------------------------------------------------------------------
// bf16 GEMM: C(MxN) = A(MxK) @ B(KxN) (+bias), fp32 or bf16 output.
// Block = 128 threads (4 waves). Tile 128(M) x 64(N) x 32(K). 8 WMMAs/wave/kstep.
// Preconditions (all call sites satisfy): M%128==0, N%64==0, K%32==0,
// lda%8==0, ldb%8==0, A/B 16-byte aligned.
// ---------------------------------------------------------------------------
__global__ __launch_bounds__(128) void gemm_bf16_kernel(
    const u16* __restrict__ A, const u16* __restrict__ B,
    const float* __restrict__ bias, void* __restrict__ Cout,
    int K, int lda, int ldb, int ldc, int out_bf16)
{
  __shared__ u16 As[128][40];  // rows padded: 80B stride, 16B-aligned chunks
  __shared__ u16 Bt[64][40];   // transposed B tile: Bt[n][k]
  const int tid = threadIdx.x;
  const int lane = tid & 31;
  const int wv = tid >> 5;
  const size_t m0 = (size_t)blockIdx.y * 128;
  const size_t n0 = (size_t)blockIdx.x * 64;

  AccF acc[2][4];
#pragma unroll
  for (int a = 0; a < 2; ++a)
#pragma unroll
    for (int b = 0; b < 4; ++b)
#pragma unroll
      for (int r = 0; r < 8; ++r) acc[a][b].f[r] = 0.0f;

  const int kb = (lane >= 16) ? 16 : 0;
  const int kh = (lane >= 16) ? 16 : 0;
  const int l15 = lane & 15;

  const int ktiles = K >> 5;
  for (int t = 0; t < ktiles; ++t) {
    const int k0 = t << 5;
    // A tile 128x32: 512 b128 loads (4/thread)
#pragma unroll
    for (int i = 0; i < 4; ++i) {
      int idx = tid + i * 128;
      int r = idx >> 2, c8 = (idx & 3) << 3;
      *(uint4*)&As[r][c8] = *(const uint4*)(A + (m0 + r) * lda + k0 + c8);
    }
    // B tile 32x64 -> Bt[n][k]: 256 b128 loads (2/thread), transpose via b16 stores
#pragma unroll
    for (int i = 0; i < 2; ++i) {
      int idx = tid + i * 128;
      int r = idx >> 3, c8 = (idx & 7) << 3;
      U4 bv;
      bv.q = *(const uint4*)(B + (size_t)(k0 + r) * ldb + n0 + c8);
#pragma unroll
      for (int j = 0; j < 8; ++j) Bt[c8 + j][r] = bv.h[j];
    }
    __syncthreads();

    BFrag af[2];
#pragma unroll
    for (int mt = 0; mt < 2; ++mt)
      lda_frag(af[mt], &As[(wv * 2 + mt) * 16 + l15][0], kb);
#pragma unroll
    for (int nt = 0; nt < 4; ++nt) {
      BFrag bfr;
      ldb_frag(bfr, &Bt[nt * 16 + l15][0], kh);
#pragma unroll
      for (int mt = 0; mt < 2; ++mt)
        acc[mt][nt].v = __builtin_amdgcn_wmma_f32_16x16x32_bf16(
            false, af[mt].v, false, bfr.v, (short)0, acc[mt][nt].v, false, false);
    }
    __syncthreads();
  }

#pragma unroll
  for (int nt = 0; nt < 4; ++nt) {
    const size_t n = n0 + (nt << 4) + l15;
    const float bv = bias ? bias[n] : 0.0f;
#pragma unroll
    for (int mt = 0; mt < 2; ++mt) {
      const size_t mb = m0 + (wv * 2 + mt) * 16 + ((lane >= 16) ? 8 : 0);
#pragma unroll
      for (int r = 0; r < 8; ++r) {
        float v = acc[mt][nt].f[r] + bv;
        if (out_bf16) ((u16*)Cout)[(mb + r) * ldc + n] = f2bf(v);
        else          ((float*)Cout)[(mb + r) * ldc + n] = v;
      }
    }
  }
}

// ---------------------------------------------------------------------------
// Per-edge Wigner rotation: mm[j,:] = sum_n wig[e,PM[j],n] * [x[src]|x[tgt]](n,:)
// then scale by rad and emit bf16.  M=29(pad32), K=49(pad64), N=256. 8 waves.
// ---------------------------------------------------------------------------
__global__ __launch_bounds__(256) void rot_fwd_kernel(
    const u16* __restrict__ wig, const u16* __restrict__ xbf,
    const u16* __restrict__ radbf, const int* __restrict__ eidx,
    u16* __restrict__ mmrad)
{
  __shared__ u16 As[32][72];    // 29x49 padded; 144B row stride
  __shared__ u16 Bt[256][72];   // transposed [x_src|x_tgt]: Bt[c][n]
  const int e = blockIdx.x;
  const int tid = threadIdx.x, lane = tid & 31, wv = tid >> 5;
  const int src = eidx[e], tgt = eidx[NE + e];
  const u16* wge = wig + (size_t)e * (MRED * NCH);

  for (int i = tid; i < 32 * 32; i += 256) {
    int j = i >> 5, k2 = (i & 31) << 1;
    u32 v = 0;
    if (j < MRED) {
      u32 a = (k2 < NCH)     ? wge[PM[j] * NCH + k2]     : 0u;
      u32 b = (k2 + 1 < NCH) ? wge[PM[j] * NCH + k2 + 1] : 0u;
      v = a | (b << 16);
    }
    *(u32*)&As[j][k2] = v;
  }
  // B tile 64x256: b128 global loads along c, transpose into Bt
#pragma unroll
  for (int i = 0; i < 8; ++i) {
    int idx = tid + i * 256;
    int n = idx >> 5, c8 = (idx & 31) << 3;
    U4 bv;
    bv.q = make_uint4(0u, 0u, 0u, 0u);
    if (n < NCH) {
      const u16* xr = (c8 < CCH) ? (xbf + ((size_t)src * NCH + n) * CCH + c8)
                                 : (xbf + ((size_t)tgt * NCH + n) * CCH + (c8 - CCH));
      bv.q = *(const uint4*)xr;
    }
#pragma unroll
    for (int j = 0; j < 8; ++j) Bt[c8 + j][n] = bv.h[j];
  }
  __syncthreads();

  AccF acc[2][2];
#pragma unroll
  for (int a = 0; a < 2; ++a)
#pragma unroll
    for (int b = 0; b < 2; ++b)
#pragma unroll
      for (int r = 0; r < 8; ++r) acc[a][b].f[r] = 0.0f;

  const int kb = (lane >= 16) ? 16 : 0;
  const int kh = (lane >= 16) ? 16 : 0;
  const int l15 = lane & 15;

  for (int k0 = 0; k0 < 64; k0 += 32) {
    BFrag af[2];
#pragma unroll
    for (int mt = 0; mt < 2; ++mt)
      lda_frag(af[mt], &As[(mt << 4) + l15][0], k0 + kb);
#pragma unroll
    for (int nt = 0; nt < 2; ++nt) {
      const int gn = ((wv * 2 + nt) << 4) + l15;
      BFrag bfr;
      ldb_frag(bfr, &Bt[gn][0], k0 + kh);
#pragma unroll
      for (int mt = 0; mt < 2; ++mt)
        acc[mt][nt].v = __builtin_amdgcn_wmma_f32_16x16x32_bf16(
            false, af[mt].v, false, bfr.v, (short)0, acc[mt][nt].v, false, false);
    }
  }

  const u16* rade = radbf + (size_t)e * RADTOT;
  u16* mme = mmrad + (size_t)e * (MRED * 256);
#pragma unroll
  for (int mt = 0; mt < 2; ++mt)
#pragma unroll
    for (int nt = 0; nt < 2; ++nt) {
      const int n = ((wv * 2 + nt) << 4) + l15;
#pragma unroll
      for (int r = 0; r < 8; ++r) {
        const int j = (mt << 4) + ((lane >= 16) ? 8 : 0) + r;
        if (j < MRED) {
          int radix = (j < 7)  ? j * 256 + n
                    : (j < 19) ? 1792 + ((j - 7) % 6) * 256 + n
                               : 3328 + ((j - 19) % 5) * 256 + n;
          mme[j * 256 + n] = f2bf(acc[mt][nt].f[r] * bf2f(rade[radix]));
        }
      }
    }
}

// ---------------------------------------------------------------------------
// Per-edge out_e = wig^T (49x29) @ val (29x128), atomicAdd into node[tgt].
// ---------------------------------------------------------------------------
__global__ __launch_bounds__(256) void out_e_kernel(
    const u16* __restrict__ wig, const u16* __restrict__ valbf,
    const int* __restrict__ eidx, float* __restrict__ node)
{
  __shared__ u16 As[64][40];    // wig^T: 49x29 padded
  __shared__ u16 Bt[128][40];   // transposed val tile: Bt[c][k]
  const int e = blockIdx.x;
  const int tid = threadIdx.x, lane = tid & 31, wv = tid >> 5;
  const int tgt = eidx[NE + e];
  const u16* wge = wig + (size_t)e * (MRED * NCH);

  for (int i = tid; i < 64 * 16; i += 256) {
    int m = i >> 4, k2 = (i & 15) << 1;
    u32 v = 0;
    if (m < NCH) {
      u32 a = (k2 < MRED)     ? wge[k2 * NCH + m]       : 0u;
      u32 b = (k2 + 1 < MRED) ? wge[(k2 + 1) * NCH + m] : 0u;
      v = a | (b << 16);
    }
    *(u32*)&As[m][k2] = v;
  }
  // B tile 32x128: b128 loads along c, transpose into Bt
#pragma unroll
  for (int i = 0; i < 2; ++i) {
    int idx = tid + i * 256;
    int k = idx >> 4, c8 = (idx & 15) << 3;
    U4 bv;
    bv.q = make_uint4(0u, 0u, 0u, 0u);
    if (k < MRED)
      bv.q = *(const uint4*)(valbf + (size_t)e * (MRED * VV) + k * VV + c8);
#pragma unroll
    for (int j = 0; j < 8; ++j) Bt[c8 + j][k] = bv.h[j];
  }
  __syncthreads();

  const int kb = (lane >= 16) ? 16 : 0;
  const int kh = (lane >= 16) ? 16 : 0;
  const int l15 = lane & 15;
  const int gn = (wv << 4) + l15;

  BFrag bfr;
  ldb_frag(bfr, &Bt[gn][0], kh);

  AccF acc[4];
#pragma unroll
  for (int a = 0; a < 4; ++a)
#pragma unroll
    for (int r = 0; r < 8; ++r) acc[a].f[r] = 0.0f;

#pragma unroll
  for (int mt = 0; mt < 4; ++mt) {
    BFrag af;
    lda_frag(af, &As[(mt << 4) + l15][0], kb);
    acc[mt].v = __builtin_amdgcn_wmma_f32_16x16x32_bf16(
        false, af.v, false, bfr.v, (short)0, acc[mt].v, false, false);
  }

  float* nb = node + (size_t)tgt * (NCH * VV);
#pragma unroll
  for (int mt = 0; mt < 4; ++mt)
#pragma unroll
    for (int r = 0; r < 8; ++r) {
      const int m = (mt << 4) + ((lane >= 16) ? 8 : 0) + r;
      if (m < NCH) atomicAdd(&nb[m * VV + gn], acc[mt].f[r]);
    }
}

// ---------------------------------------------------------------------------
// Elementwise / reduction kernels
// ---------------------------------------------------------------------------
__global__ void zero_kernel(u32* p, long n) {
  long i = (long)blockIdx.x * 256 + threadIdx.x;
  if (i < n) p[i] = 0u;
}
// float4 -> 4x bf16 packed u64; n4 = count/4
__global__ void conv4_bf_kernel(const float* __restrict__ in, u16* __restrict__ out, long n4) {
  long i = (long)blockIdx.x * 256 + threadIdx.x;
  if (i >= n4) return;
  float4 v = ((const float4*)in)[i];
  u64 p = (u64)f2bf(v.x) | ((u64)f2bf(v.y) << 16) | ((u64)f2bf(v.z) << 32) | ((u64)f2bf(v.w) << 48);
  ((u64*)out)[i] = p;
}
__global__ void projT_kernel(const float* pw, u16* pt) {
  int i = blockIdx.x * 256 + threadIdx.x;
  if (i >= 7 * 128 * 128) return;
  int l = i >> 14, o = (i >> 7) & 127, k = i & 127;
  pt[(l << 14) + (k << 7) + o] = f2bf(pw[i]);
}
__global__ void gather_xedge_kernel(const float* ed, const int* an, const int* eidx,
                                    const float* semb, const float* temb, u16* xe)
{
  long i = (long)blockIdx.x * 256 + threadIdx.x;
  if (i >= (long)NE * 768) return;
  int e = (int)(i / 768), c = (int)(i % 768);
  float v;
  if (c < 512)      v = ed[(size_t)e * 512 + c];
  else if (c < 640) v = semb[an[eidx[e]] * 128 + (c - 512)];
  else              v = temb[an[eidx[NE + e]] * 128 + (c - 640)];
  xe[i] = f2bf(v);
}
__global__ __launch_bounds__(128) void ln_silu_kernel(
    const float* X, const float* g, const float* bt, u16* out)
{
  __shared__ float red[128];
  const int row = blockIdx.x, tid = threadIdx.x;
  float v = X[(size_t)row * 128 + tid];
  red[tid] = v; __syncthreads();
  for (int s = 64; s > 0; s >>= 1) { if (tid < s) red[tid] += red[tid + s]; __syncthreads(); }
  float mu = red[0] * (1.0f / 128.0f); __syncthreads();
  float d = v - mu;
  red[tid] = d * d; __syncthreads();
  for (int s = 64; s > 0; s >>= 1) { if (tid < s) red[tid] += red[tid + s]; __syncthreads(); }
  float var = red[0] * (1.0f / 128.0f);
  float y = d * rsqrtf(var + 1e-5f) * g[tid] + bt[tid];
  out[(size_t)row * 128 + tid] = f2bf(siluf(y));
}
__global__ void combine1_kernel(const float* x0full, const float* y1, const float* y2,
                                float* concatR)
{
  long i = (long)blockIdx.x * 256 + threadIdx.x;
  if (i >= (long)NE * MRED * HH) return;
  int c = (int)(i & 63), j = (int)((i >> 6) % 29), e = (int)(i / (29 * 64));
  float v;
  if (j < 7)       v = x0full[(size_t)e * X0N + EXTRAC + j * 64 + c];
  else if (j < 13) { int q = j - 7;  v = y1[(size_t)e*1536 + q*64+c]       - y1[(size_t)e*1536 + 1152 + q*64+c]; }
  else if (j < 19) { int q = j - 13; v = y1[(size_t)e*1536 + 768 + q*64+c] + y1[(size_t)e*1536 + 384 + q*64+c]; }
  else if (j < 24) { int q = j - 19; v = y2[(size_t)e*1280 + q*64+c]       - y2[(size_t)e*1280 + 960 + q*64+c]; }
  else             { int q = j - 24; v = y2[(size_t)e*1280 + 640 + q*64+c] + y2[(size_t)e*1280 + 320 + q*64+c]; }
  concatR[(size_t)e * 1856 + j * 64 + c] = v;
}
__global__ __launch_bounds__(256) void att_logit_kernel(
    const float* x0full, const float* ag, const float* ab, const float* adot, float* logit)
{
  const int e = blockIdx.x, tid = threadIdx.x, lane = tid & 31, h = tid >> 5;
  const float* af = x0full + (size_t)e * X0N + h * 64;
  float a0 = af[lane], a1 = af[lane + 32];
  float s = a0 + a1;
  for (int i = 16; i > 0; i >>= 1) s += __shfl_xor(s, i, 32);
  float mu = s * (1.0f / 64.0f);
  float d0 = a0 - mu, d1 = a1 - mu;
  float q = d0 * d0 + d1 * d1;
  for (int i = 16; i > 0; i >>= 1) q += __shfl_xor(q, i, 32);
  float inv = rsqrtf(q * (1.0f / 64.0f) + 1e-5f);
  float n0 = d0 * inv * ag[lane] + ab[lane];
  float n1 = d1 * inv * ag[lane + 32] + ab[lane + 32];
  float s0 = 1.0f / (1.0f + __expf(-n0));
  float s1 = 1.0f / (1.0f + __expf(-n1));
  float t0 = 0.6f * n0 + 0.4f * n0 * (2.0f * s0 - 1.0f);
  float t1 = 0.6f * n1 + 0.4f * n1 * (2.0f * s1 - 1.0f);
  float t = t0 * adot[h * 64 + lane] + t1 * adot[h * 64 + lane + 32];
  for (int i = 16; i > 0; i >>= 1) t += __shfl_xor(t, i, 32);
  if (lane == 0) logit[e * 8 + h] = t;
}
__device__ __forceinline__ u32 fenc(float f) {
  u32 u = __float_as_uint(f);
  return (u & 0x80000000u) ? ~u : (u | 0x80000000u);
}
__device__ __forceinline__ float fdec(u32 e) {
  return __uint_as_float((e & 0x80000000u) ? (e & 0x7FFFFFFFu) : ~e);
}
__global__ void att_max_kernel(const float* logit, const int* eidx, u32* amax) {
  int i = blockIdx.x * 256 + threadIdx.x;
  if (i >= NE * 8) return;
  int e = i >> 3, h = i & 7;
  atomicMax(&amax[eidx[NE + e] * 8 + h], fenc(logit[i]));
}
__global__ void att_exp_kernel(const float* logit, const int* eidx, const u32* amax,
                               float* ex, float* den) {
  int i = blockIdx.x * 256 + threadIdx.x;
  if (i >= NE * 8) return;
  int e = i >> 3, h = i & 7, tgt = eidx[NE + e];
  float v = __expf(logit[i] - fdec(amax[tgt * 8 + h]));
  ex[i] = v;
  atomicAdd(&den[tgt * 8 + h], v);
}
__global__ __launch_bounds__(256) void grid_kernel(
    const float* concatR, const float* x0full, const float* Gg, const float* Fg, u16* mm2)
{
  __shared__ float msg2s[29][64];
  __shared__ float tacc[4][29][64];
  const int e = blockIdx.x, tid = threadIdx.x;
  for (int i = tid; i < 29 * 64; i += 256) {
    int j = i >> 6, c = i & 63;
    msg2s[j][c] = concatR[(size_t)e * 1856 + PL[j] * 64 + c];
  }
  __syncthreads();
  const int c = tid & 63, grp = tid >> 6;
  float part[29];
  for (int i = 0; i < 29; ++i) part[i] = 0.0f;
  for (int b = grp; b < 196; b += 4) {
    const float* gr = Gg + b * 29;
    float s = 0.0f;
    for (int i = 0; i < 29; ++i) s += gr[i] * msg2s[i][c];
    float gv = siluf(s);
    const float* fr = Fg + b * 29;
    for (int i = 0; i < 29; ++i) part[i] += fr[i] * gv;
  }
  for (int i = 0; i < 29; ++i) tacc[grp][i][c] = part[i];
  __syncthreads();
  for (int i = tid; i < 29 * 64; i += 256) {
    int j = i >> 6, cc = i & 63;
    int srow = PM[j];
    float v = (srow == 0)
                  ? siluf(x0full[(size_t)e * X0N + ALPHACH + cc])
                  : (tacc[0][srow][cc] + tacc[1][srow][cc] + tacc[2][srow][cc] + tacc[3][srow][cc]);
    mm2[(size_t)e * 1856 + j * 64 + cc] = f2bf(v);
  }
}
__global__ void combine_v_kernel(const float* x0v, const float* y1v, const float* y2v,
                                 const float* ex, const float* den, const int* eidx,
                                 u16* valbf)
{
  long i = (long)blockIdx.x * 256 + threadIdx.x;
  if (i >= (long)NE * MRED * VV) return;
  int c = (int)(i & 127), j = (int)((i >> 7) % 29), e = (int)(i / (29 * 128));
  float v;
  if (j < 7)       v = x0v[(size_t)e * 896 + j * 128 + c];
  else if (j < 13) { int q = j - 7;  v = y1v[(size_t)e*3072 + q*128+c]        - y1v[(size_t)e*3072 + 2304 + q*128+c]; }
  else if (j < 19) { int q = j - 13; v = y1v[(size_t)e*3072 + 1536 + q*128+c] + y1v[(size_t)e*3072 + 768 + q*128+c]; }
  else if (j < 24) { int q = j - 19; v = y2v[(size_t)e*2560 + q*128+c]        - y2v[(size_t)e*2560 + 1920 + q*128+c]; }
  else             { int q = j - 24; v = y2v[(size_t)e*2560 + 1280 + q*128+c] + y2v[(size_t)e*2560 + 640 + q*128+c]; }
  int h = c >> 4;
  int tgt = eidx[NE + e];
  float alpha = ex[e * 8 + h] / (den[tgt * 8 + h] + 1e-16f);
  valbf[(size_t)e * 3712 + PM[j] * 128 + c] = f2bf(v * alpha);
}

// ---------------------------------------------------------------------------
// Host launcher
// ---------------------------------------------------------------------------
extern "C" void kernel_launch(void* const* d_in, const int* in_sizes, int n_in,
                              void* d_out, int out_size, void* d_ws, size_t ws_size,
                              hipStream_t stream)
{
  (void)in_sizes; (void)n_in; (void)out_size; (void)ws_size;
  const float* x        = (const float*)d_in[0];
  const int*   an       = (const int*)d_in[1];
  const float* edist    = (const float*)d_in[2];
  const int*   eidx     = (const int*)d_in[3];
  const float* semb     = (const float*)d_in[4];
  const float* temb     = (const float*)d_in[5];
  const float* rad_w1   = (const float*)d_in[6];
  const float* rad_b1   = (const float*)d_in[7];
  const float* rad_g1   = (const float*)d_in[8];
  const float* rad_bt1  = (const float*)d_in[9];
  const float* rad_w2   = (const float*)d_in[10];
  const float* rad_b2   = (const float*)d_in[11];
  const float* rad_g2   = (const float*)d_in[12];
  const float* rad_bt2  = (const float*)d_in[13];
  const float* rad_w3   = (const float*)d_in[14];
  const float* rad_b3   = (const float*)d_in[15];
  const float* wigner   = (const float*)d_in[16];
  const float* c1m0w    = (const float*)d_in[17];
  const float* c1m0b    = (const float*)d_in[18];
  const float* c1m1w    = (const float*)d_in[19];
  const float* c1m2w    = (const float*)d_in[20];
  const float* alpha_g  = (const float*)d_in[21];
  const float* alpha_b  = (const float*)d_in[22];
  const float* alpha_dt = (const float*)d_in[23];
  const float* to_grid  = (const float*)d_in[24];
  const float* from_grid= (const float*)d_in[25];
  const float* c2m0w    = (const float*)d_in[26];
  const float* c2m0b    = (const float*)d_in[27];
  const float* c2m1w    = (const float*)d_in[28];
  const float* c2m2w    = (const float*)d_in[29];
  const float* proj_w   = (const float*)d_in[30];
  const float* proj_b   = (const float*)d_in[31];

  char* ws = (char*)d_ws;
  size_t off = 0;
  auto alloc = [&](size_t bytes) -> char* {
    char* p = ws + off;
    off = (off + bytes + 255) & ~(size_t)255;
    return p;
  };

  u16*   x_bf    = (u16*)alloc((size_t)NNODE * 49 * 128 * 2);
  char*  arena1  = alloc(12845056);            // xedge_bf (S1-2) | node_bf (S9)
  u16*   xedge_bf= (u16*)arena1;
  u16*   node_bf = (u16*)arena1;
  float* hf32    = (float*)alloc((size_t)NE * 128 * 4);
  u16*   h1_bf   = (u16*)alloc((size_t)NE * 128 * 2);
  u16*   h2_bf   = (u16*)alloc((size_t)NE * 128 * 2);
  char*  arenaA  = alloc(100663296);           // rad_bf (S3-5) | y1v (S8)
  u16*   rad_bf  = (u16*)arenaA;
  float* y1v     = (float*)arenaA;
  u16*   wig_bf  = (u16*)alloc((size_t)NE * MRED * NCH * 2);
  char*  arenaB  = alloc(121634816);           // mmrad (S5-6) | y2v (S8)
  u16*   mmrad   = (u16*)arenaB;
  float* y2v     = (float*)arenaB;
  float* x0full  = (float*)alloc((size_t)NE * X0N * 4);
  char*  arenaC  = alloc(60817408);            // y1 (S6) | val_bf (S8-9)
  float* y1      = (float*)arenaC;
  u16*   val_bf  = (u16*)arenaC;
  char*  arenaD  = alloc(41943040);            // y2 (S6) | x0v (S8)
  float* y2      = (float*)arenaD;
  float* x0v     = (float*)arenaD;
  float* concatR = (float*)alloc((size_t)NE * 1856 * 4);
  u16*   mm2_bf  = (u16*)alloc((size_t)NE * 1856 * 2);
  float* logit   = (float*)alloc((size_t)NE * 8 * 4);
  float* exb     = (float*)alloc((size_t)NE * 8 * 4);
  u32*   amax    = (u32*)alloc((size_t)NNODE * 8 * 4);
  float* den     = (float*)alloc((size_t)NNODE * 8 * 4);
  float* node    = (float*)alloc((size_t)NNODE * 49 * 128 * 4);
  u16*   w1bf    = (u16*)alloc(768 * 128 * 2);
  u16*   w2bf    = (u16*)alloc(128 * 128 * 2);
  u16*   w3bf    = (u16*)alloc(128 * 4608 * 2);
  u16*   c1m0bf  = (u16*)alloc((size_t)1792 * 1024 * 2);
  u16*   c1m1bf  = (u16*)alloc((size_t)1536 * 768 * 2);
  u16*   c1m2bf  = (u16*)alloc((size_t)1280 * 640 * 2);
  u16*   c2m0bf  = (u16*)alloc((size_t)448 * 896 * 2);
  u16*   c2m1bf  = (u16*)alloc((size_t)384 * 1536 * 2);
  u16*   c2m2bf  = (u16*)alloc((size_t)320 * 1280 * 2);
  u16*   projTbf = (u16*)alloc((size_t)7 * 128 * 128 * 2);

  auto blks = [](long n) { return dim3((unsigned)((n + 255) / 256)); };
  const dim3 t256(256);

  // init accumulators
  zero_kernel<<<blks((long)NNODE * 49 * 128), t256, 0, stream>>>((u32*)node, (long)NNODE * 49 * 128);
  zero_kernel<<<blks(NNODE * 8), t256, 0, stream>>>(amax, NNODE * 8);
  zero_kernel<<<blks(NNODE * 8), t256, 0, stream>>>((u32*)den, NNODE * 8);

  // bf16 conversions (all counts are multiples of 4)
  auto conv4 = [&](const float* in, u16* out, long n) {
    conv4_bf_kernel<<<blks(n / 4), t256, 0, stream>>>(in, out, n / 4);
  };
  conv4(x, x_bf, (long)NNODE * 49 * 128);
  conv4(wigner, wig_bf, (long)NE * MRED * NCH);
  conv4(rad_w1, w1bf, 768 * 128);
  conv4(rad_w2, w2bf, 128 * 128);
  conv4(rad_w3, w3bf, 128 * 4608);
  conv4(c1m0w, c1m0bf, (long)1792 * 1024);
  conv4(c1m1w, c1m1bf, (long)1536 * 768);
  conv4(c1m2w, c1m2bf, (long)1280 * 640);
  conv4(c2m0w, c2m0bf, (long)448 * 896);
  conv4(c2m1w, c2m1bf, (long)384 * 1536);
  conv4(c2m2w, c2m2bf, (long)320 * 1280);
  projT_kernel<<<blks(7 * 128 * 128), t256, 0, stream>>>(proj_w, projTbf);
  gather_xedge_kernel<<<blks((long)NE * 768), t256, 0, stream>>>(edist, an, eidx, semb, temb, xedge_bf);

  auto gemm = [&](const u16* A, const u16* B, const float* bias, void* C,
                  int M, int Np, int K, int lda, int ldb, int ldc, int obf) {
    dim3 g((unsigned)(Np / 64), (unsigned)(M / 128));
    gemm_bf16_kernel<<<g, dim3(128), 0, stream>>>(A, B, bias, C, K, lda, ldb, ldc, obf);
  };

  // radial MLP
  gemm(xedge_bf, w1bf, rad_b1, hf32, NE, 128, 768, 768, 128, 128, 0);
  ln_silu_kernel<<<NE, 128, 0, stream>>>(hf32, rad_g1, rad_bt1, h1_bf);
  gemm(h1_bf, w2bf, rad_b2, hf32, NE, 128, 128, 128, 128, 128, 0);
  ln_silu_kernel<<<NE, 128, 0, stream>>>(hf32, rad_g2, rad_bt2, h2_bf);
  gemm(h2_bf, w3bf, rad_b3, rad_bf, NE, 4608, 128, 128, 4608, 4608, 1);

  // Wigner rotate + rad scale (PERM_M fused)
  rot_fwd_kernel<<<NE, t256, 0, stream>>>(wig_bf, x_bf, rad_bf, eidx, mmrad);

  // SO2 conv #1
  gemm(mmrad, c1m0bf, c1m0b, x0full, NE, 1024, 1792, 7424, 1024, 1024, 0);
  for (int r = 0; r < 2; ++r)
    gemm(mmrad + 1792 + r * 1536, c1m1bf, nullptr, y1 + r * 768, NE, 768, 1536, 7424, 768, 1536, 0);
  for (int r = 0; r < 2; ++r)
    gemm(mmrad + 4864 + r * 1280, c1m2bf, nullptr, y2 + r * 640, NE, 640, 1280, 7424, 640, 1280, 0);
  combine1_kernel<<<blks((long)NE * MRED * HH), t256, 0, stream>>>(x0full, y1, y2, concatR);

  // attention softmax over segments
  att_logit_kernel<<<NE, t256, 0, stream>>>(x0full, alpha_g, alpha_b, alpha_dt, logit);
  att_max_kernel<<<blks(NE * 8), t256, 0, stream>>>(logit, eidx, amax);
  att_exp_kernel<<<blks(NE * 8), t256, 0, stream>>>(logit, eidx, amax, exb, den);

  // grid nonlinearity (PERM_L gather + PERM_M scatter fused)
  grid_kernel<<<NE, t256, 0, stream>>>(concatR, x0full, to_grid, from_grid, mm2_bf);

  // SO2 conv #2
  gemm(mm2_bf, c2m0bf, c2m0b, x0v, NE, 896, 448, 1856, 896, 896, 0);
  for (int r = 0; r < 2; ++r)
    gemm(mm2_bf + 448 + r * 384, c2m1bf, nullptr, y1v + r * 1536, NE, 1536, 384, 1856, 1536, 3072, 0);
  for (int r = 0; r < 2; ++r)
    gemm(mm2_bf + 1216 + r * 320, c2m2bf, nullptr, y2v + r * 1280, NE, 1280, 320, 1856, 1280, 2560, 0);
  combine_v_kernel<<<blks((long)NE * MRED * VV), t256, 0, stream>>>(x0v, y1v, y2v, exb, den, eidx, val_bf);

  // inverse Wigner + segment-sum into nodes
  out_e_kernel<<<NE, t256, 0, stream>>>(wig_bf, val_bf, eidx, node);

  // final per-l projection (49 strided GEMMs)
  conv4(node, node_bf, (long)NNODE * 49 * 128);
  for (int m = 0; m < 49; ++m) {
    int l = 0;
    while ((l + 1) * (l + 1) <= m) ++l;
    gemm(node_bf + m * 128, projTbf + l * 16384, (m == 0) ? proj_b : nullptr,
         (float*)d_out + m * 128, NNODE, 128, 128, 49 * 128, 128, 49 * 128, 0);
  }
}
